// RegionLearner_35218731828201
// MI455X (gfx1250) — matrix operand order repaired and checked by reading, not compiled, not run on hardware
//
#include <hip/hip_runtime.h>
#include <hip/hip_bf16.h>
#include <stdint.h>

typedef __attribute__((ext_vector_type(16))) __bf16 v16bf;
typedef __attribute__((ext_vector_type(8)))  float  v8f;

#define B_    64
#define L_    196
#define C_    768
#define K_    8192
#define R_    8
#define H_    14
#define F_    4
#define BP_   16
#define HEADS 8
#define DH    96
#define SEQ   32   /* cur_f * R */

union BF16Frag { v16bf v; uint4 q[2]; };

static __device__ __forceinline__ __bf16 f2bf(float f) {
  unsigned u = __float_as_uint(f);
  unsigned r = (u + 0x7FFFu + ((u >> 16) & 1u)) >> 16;   // round-nearest-even
  unsigned short s = (unsigned short)r;
  union { unsigned short u16; __bf16 h; } cv; cv.u16 = s;
  return cv.h;
}

/* ---------------- prep: fp32 -> bf16 casts + codebook norms ---------------- */

__global__ __launch_bounds__(256) void k_cvt_x(const float* __restrict__ x,
                                               __bf16* __restrict__ xb, int n) {
  int i = blockIdx.x * 256 + threadIdx.x;
  if (i < n) xb[i] = f2bf(x[i]);
}

__global__ __launch_bounds__(256) void k_prep_cb(const float* __restrict__ c,
                                                 __bf16* __restrict__ cb,
                                                 float* __restrict__ cn) {
  __shared__ float red[256];
  int row = blockIdx.x, tid = threadIdx.x;
  const float* src = c + (size_t)row * C_;
  float acc = 0.f;
  for (int j = tid; j < C_; j += 256) {
    float f = src[j];
    cb[(size_t)row * C_ + j] = f2bf(f);
    acc += f * f;
  }
  red[tid] = acc; __syncthreads();
  for (int s = 128; s > 0; s >>= 1) { if (tid < s) red[tid] += red[tid + s]; __syncthreads(); }
  if (tid == 0) cn[row] = red[0];
}

/* ------------- VQ: WMMA bf16 GEMM + fused argmin + codebook gather ---------
   grid = 392 row-tiles of 32; block = 256 (8 waves).
   Wave computes a 32x32 output patch: 2 row-tiles (A from 48KB LDS tile)
   x 2 col-stripes (B streamed from L2-resident codebook) = 4 accumulators,
   so each global B fragment feeds 2 WMMAs and each LDS A fragment feeds 2.
   Next k-step fragments are preloaded before current WMMAs (sw pipeline).
   Codebook traffic: 392 x 12.6MB ~= 5 GB of L2 reads for the whole GEMM.  */

__global__ __launch_bounds__(256) void k_vq_argmin(
    const __bf16* __restrict__ xb, const __bf16* __restrict__ cb,
    const float* __restrict__ cn, const float* __restrict__ codebook,
    float* __restrict__ vq, int* __restrict__ enc)
{
  __shared__ __align__(16) __bf16 lx[32 * C_];
  __shared__ unsigned long long red[32];
  int tid  = threadIdx.x;
  int tile = blockIdx.x;

  { /* stage 32x768 bf16 X tile into LDS (48 KB) */
    const uint4* src = (const uint4*)(xb + (size_t)tile * 32 * C_);
    uint4* dst = (uint4*)lx;
    for (int i = tid; i < (32 * C_) / 8; i += 256) dst[i] = src[i];
  }
  if (tid < 32) red[tid] = 0xFFFFFFFFFFFFFFFFull;
  __syncthreads();

  int w = tid >> 5, lane = tid & 31;
  int lo = lane & 15, hi = lane >> 4;
  const int arow0 = lo * C_ + hi * 16;           /* A rows 0..15  */
  const int arow1 = (lo + 16) * C_ + hi * 16;    /* A rows 16..31 */

  /* running min per output row half; both col-stripes fold into the same row */
  float msL[8], msH[8]; int miL[8], miH[8];
#pragma unroll
  for (int r = 0; r < 8; ++r) { msL[r] = 3.4e38f; miL[r] = 0; msH[r] = 3.4e38f; miH[r] = 0; }

  for (int chunk = 0; chunk < 32; ++chunk) {
    int n0 = chunk * 256 + w * 32 + lo;         /* this lane's two columns */
    int n1 = n0 + 16;
    const __bf16* b0row = cb + (size_t)n0 * C_;
    const __bf16* b1row = cb + (size_t)n1 * C_;
    if (chunk + 1 < 32) {                       /* hint next chunk into caches */
      __builtin_prefetch(b0row + (size_t)256 * C_, 0, 1);
      __builtin_prefetch(b1row + (size_t)256 * C_, 0, 1);
    }

    v8f acc00 = {}, acc01 = {}, acc10 = {}, acc11 = {};
    BF16Frag a0, a1, b0, b1, a0n, a1n, b0n, b1n;

    /* prologue: k-step 0 fragments */
    a0.q[0] = *(const uint4*)&lx[arow0];
    a0.q[1] = *(const uint4*)&lx[arow0 + 8];
    a1.q[0] = *(const uint4*)&lx[arow1];
    a1.q[1] = *(const uint4*)&lx[arow1 + 8];
    { const uint4* p0 = (const uint4*)(b0row + hi * 16);
      const uint4* p1 = (const uint4*)(b1row + hi * 16);
      b0.q[0] = p0[0]; b0.q[1] = p0[1];
      b1.q[0] = p1[0]; b1.q[1] = p1[1]; }

#pragma unroll
    for (int kb = 0; kb < 24; ++kb) {           /* K = 768 = 24*32 */
      if (kb < 23) {                            /* preload k-step kb+1 */
        int kn = (kb + 1) * 32;
        a0n.q[0] = *(const uint4*)&lx[arow0 + kn];
        a0n.q[1] = *(const uint4*)&lx[arow0 + kn + 8];
        a1n.q[0] = *(const uint4*)&lx[arow1 + kn];
        a1n.q[1] = *(const uint4*)&lx[arow1 + kn + 8];
        const uint4* p0 = (const uint4*)(b0row + kn + hi * 16);
        const uint4* p1 = (const uint4*)(b1row + kn + hi * 16);
        b0n.q[0] = p0[0]; b0n.q[1] = p0[1];
        b1n.q[0] = p1[0]; b1n.q[1] = p1[1];
      }
      acc00 = __builtin_amdgcn_wmma_f32_16x16x32_bf16(false, a0.v, false, b0.v,
                                                      (short)0, acc00, false, false);
      acc01 = __builtin_amdgcn_wmma_f32_16x16x32_bf16(false, a0.v, false, b1.v,
                                                      (short)0, acc01, false, false);
      acc10 = __builtin_amdgcn_wmma_f32_16x16x32_bf16(false, a1.v, false, b0.v,
                                                      (short)0, acc10, false, false);
      acc11 = __builtin_amdgcn_wmma_f32_16x16x32_bf16(false, a1.v, false, b1.v,
                                                      (short)0, acc11, false, false);
      a0 = a0n; a1 = a1n; b0 = b0n; b1 = b1n;   /* renames under full unroll */
    }

    float c0 = cn[n0], c1 = cn[n1];
#pragma unroll
    for (int r = 0; r < 8; ++r) {               /* D: M = r + 8*hi (+16), N = col */
      float s;
      s = c0 - 2.0f * acc00[r]; if (s < msL[r]) { msL[r] = s; miL[r] = n0; }
      s = c1 - 2.0f * acc01[r]; if (s < msL[r]) { msL[r] = s; miL[r] = n1; }
      s = c0 - 2.0f * acc10[r]; if (s < msH[r]) { msH[r] = s; miH[r] = n0; }
      s = c1 - 2.0f * acc11[r]; if (s < msH[r]) { msH[r] = s; miH[r] = n1; }
    }
  }

  /* merge: order-preserving float key || index, ds_min_u64 per row */
#pragma unroll
  for (int r = 0; r < 8; ++r) {
    unsigned uL = __float_as_uint(msL[r]);
    uL = (uL & 0x80000000u) ? ~uL : (uL | 0x80000000u);
    atomicMin(&red[r + hi * 8],
              ((unsigned long long)uL << 32) | (unsigned)miL[r]);
    unsigned uH = __float_as_uint(msH[r]);
    uH = (uH & 0x80000000u) ? ~uH : (uH | 0x80000000u);
    atomicMin(&red[16 + r + hi * 8],
              ((unsigned long long)uH << 32) | (unsigned)miH[r]);
  }
  __syncthreads();

  if (tid < 32) enc[tile * 32 + tid] = (int)(unsigned)(red[tid] & 0xFFFFFFFFull);

  /* straight-through forward == codebook gather */
  for (int r = 0; r < 32; ++r) {
    int idx = (int)(unsigned)(red[r] & 0xFFFFFFFFull);
    const float* srcr = codebook + (size_t)idx * C_;
    float* dstr = vq + ((size_t)tile * 32 + r) * C_;
    for (int c = tid; c < C_; c += 256) dstr[c] = srcr[c];
  }
}

/* ---------------- motion excitation: frame-diff 3x3 conv -> softmax -------- */

__global__ __launch_bounds__(256) void k_mm_conv(
    const float* __restrict__ x, const float* __restrict__ exw,
    const float* __restrict__ exb, float* __restrict__ mmr)
{
  __shared__ float red[256];
  int bl = blockIdx.x;
  int b = bl / L_, l = bl % L_;
  int t = b & (F_ - 1);
  int tid = threadIdx.x;
  if (t == 0) { if (tid == 0) mmr[bl] = exb[0]; return; }   /* diff == 0 */
  int y = l / H_, x0 = l % H_;
  float acc = 0.f;
  const float* cur = x + (size_t)b * L_ * C_;
  const float* prv = x + (size_t)(b - 1) * L_ * C_;
  for (int k = 0; k < 9; ++k) {
    int yy = y + k / 3 - 1, xx = x0 + k % 3 - 1;
    if (yy < 0 || yy >= H_ || xx < 0 || xx >= H_) continue;
    const float* c0 = cur + (size_t)(yy * H_ + xx) * C_;
    const float* p0 = prv + (size_t)(yy * H_ + xx) * C_;
    for (int c = tid; c < C_; c += 256) acc += (c0[c] - p0[c]) * exw[c * 9 + k];
  }
  red[tid] = acc; __syncthreads();
  for (int s = 128; s > 0; s >>= 1) { if (tid < s) red[tid] += red[tid + s]; __syncthreads(); }
  if (tid == 0) mmr[bl] = red[0] + exb[0];
}

__global__ __launch_bounds__(256) void k_soft_apply(
    const float* __restrict__ mmr, const float* __restrict__ x,
    float* __restrict__ vq)
{
  __shared__ float p[L_];
  __shared__ float red[256];
  int b = blockIdx.x, tid = threadIdx.x;
  float v = (tid < L_) ? mmr[b * L_ + tid] : -3.4e38f;
  red[tid] = v; __syncthreads();
  for (int s = 128; s > 0; s >>= 1) { if (tid < s) red[tid] = fmaxf(red[tid], red[tid + s]); __syncthreads(); }
  float mx = red[0]; __syncthreads();
  float e = (tid < L_) ? __expf(v - mx) : 0.f;
  red[tid] = e; __syncthreads();
  for (int s = 128; s > 0; s >>= 1) { if (tid < s) red[tid] += red[tid + s]; __syncthreads(); }
  float inv = 1.f / red[0];
  if (tid < L_) p[tid] = e * inv;
  __syncthreads();
  const float* xb0 = x + (size_t)b * L_ * C_;
  float* vb = vq + (size_t)b * L_ * C_;
  for (int e2 = tid; e2 < L_ * C_; e2 += 256) vb[e2] += xb0[e2] * p[e2 / C_];
}

/* ---------------- region attention conv + masked pooling ------------------- */

__global__ __launch_bounds__(256) void k_region_mask(
    const float* __restrict__ vq, const float* __restrict__ aw,
    const float* __restrict__ ab, float* __restrict__ rm)
{
  int br = blockIdx.x;
  int b = br >> 3, r = br & 7;
  int tid = threadIdx.x;
  if (tid >= L_) return;
  int y = tid / H_, x0 = tid % H_;
  float acc = ab[r];
  const float* vb = vq + (size_t)b * L_ * C_;
  const float* wr = aw + (size_t)r * C_ * 9;
  for (int k = 0; k < 9; ++k) {
    int yy = y + k / 3 - 1, xx = x0 + k % 3 - 1;
    if (yy < 0 || yy >= H_ || xx < 0 || xx >= H_) continue;
    const float* v0 = vb + (size_t)(yy * H_ + xx) * C_;
    for (int c = 0; c < C_; ++c) acc += v0[c] * wr[c * 9 + k];
  }
  rm[(size_t)br * L_ + tid] = acc;
}

__global__ __launch_bounds__(256) void k_regions(
    const float* __restrict__ vq, const float* __restrict__ rm,
    float* __restrict__ vd0)
{
  __shared__ float m[L_];
  int br = blockIdx.x;
  int b = br >> 3;
  int tid = threadIdx.x;
  if (tid < L_) m[tid] = rm[(size_t)br * L_ + tid];
  __syncthreads();
  const float* vb = vq + (size_t)b * L_ * C_;
  for (int c = tid; c < C_; c += 256) {
    float acc = 0.f;
    for (int l = 0; l < L_; ++l) acc += vb[(size_t)l * C_ + c] * m[l];
    vd0[(size_t)br * C_ + c] = acc * (1.0f / (float)L_);
  }
}

/* ---------------- region-token self-attention ------------------------------ */

__global__ __launch_bounds__(256) void k_qkv(
    const float* __restrict__ vd0, const float* __restrict__ qw,
    const float* __restrict__ qb, float* __restrict__ qkv)
{
  __shared__ float xr[C_];
  int s = blockIdx.x, tid = threadIdx.x;
  int bp = s >> 5, tok = s & 31;
  int f = tok >> 3, r = tok & 7;
  int b = bp * F_ + f;
  const float* src = vd0 + (size_t)(b * R_ + r) * C_;
  for (int c = tid; c < C_; c += 256) xr[c] = src[c];
  __syncthreads();
  for (int j = 0; j < 9; ++j) {                 /* 2304 = 9*256 outputs */
    int o = tid + j * 256;
    const float* wrow = qw + (size_t)o * C_;
    float acc = qb[o];
    for (int c = 0; c < C_; ++c) acc += xr[c] * wrow[c];
    qkv[(size_t)s * 2304 + o] = acc;
  }
}

__global__ __launch_bounds__(256) void k_attn(
    const float* __restrict__ qkv, float* __restrict__ ao)
{
  __shared__ float qs[SEQ * DH], ks[SEQ * DH], vs[SEQ * DH], sc[SEQ * SEQ];
  int blk = blockIdx.x, tid = threadIdx.x;
  int bp = blk >> 3, h = blk & 7;
  for (int e = tid; e < SEQ * DH; e += 256) {
    int t = e / DH, d = e % DH;
    size_t base = (size_t)(bp * SEQ + t) * 2304 + h * DH + d;
    qs[e] = qkv[base]; ks[e] = qkv[base + C_]; vs[e] = qkv[base + 2 * C_];
  }
  __syncthreads();
  const float scale = 0.10206207261596575f;     /* 96^-0.5 */
  for (int e = tid; e < SEQ * SEQ; e += 256) {
    int i = e >> 5, j = e & 31;
    float acc = 0.f;
    for (int d = 0; d < DH; ++d) acc += qs[i * DH + d] * ks[j * DH + d];
    sc[e] = acc * scale;
  }
  __syncthreads();
  if (tid < SEQ) {
    float mx = -3.4e38f;
    for (int j = 0; j < SEQ; ++j) mx = fmaxf(mx, sc[tid * SEQ + j]);
    float sm = 0.f;
    for (int j = 0; j < SEQ; ++j) { float e2 = __expf(sc[tid * SEQ + j] - mx); sc[tid * SEQ + j] = e2; sm += e2; }
    float inv = 1.f / sm;
    for (int j = 0; j < SEQ; ++j) sc[tid * SEQ + j] *= inv;
  }
  __syncthreads();
  for (int e = tid; e < SEQ * DH; e += 256) {
    int t = e / DH, d = e % DH;
    float acc = 0.f;
    for (int j = 0; j < SEQ; ++j) acc += sc[t * SEQ + j] * vs[j * DH + d];
    ao[(size_t)(bp * SEQ + t) * C_ + h * DH + d] = acc;
  }
}

__global__ __launch_bounds__(256) void k_proj(
    const float* __restrict__ ao, const float* __restrict__ pw,
    const float* __restrict__ pb, float* __restrict__ vd_out)
{
  __shared__ float ar[C_];
  int s = blockIdx.x, tid = threadIdx.x;
  int bp = s >> 5, tok = s & 31;
  int f = tok >> 3, r = tok & 7;
  int b = bp * F_ + f;
  const float* src = ao + (size_t)s * C_;
  for (int c = tid; c < C_; c += 256) ar[c] = src[c];
  __syncthreads();
  for (int j = 0; j < 3; ++j) {
    int c = tid + j * 256;
    const float* wrow = pw + (size_t)c * C_;
    float acc = pb[c];
    for (int k = 0; k < C_; ++k) acc += ar[k] * wrow[k];
    vd_out[(size_t)(b * R_ + r) * C_ + c] = acc;
  }
}

/* --------------------------------- launcher -------------------------------- */

extern "C" void kernel_launch(void* const* d_in, const int* in_sizes, int n_in,
                              void* d_out, int out_size, void* d_ws, size_t ws_size,
                              hipStream_t stream) {
  (void)in_sizes; (void)n_in; (void)out_size; (void)ws_size;
  const float* in_feas  = (const float*)d_in[0];
  /* d_in[1] = cur_f (== 4, fixed by setup) */
  const float* codebook = (const float*)d_in[2];
  const float* att_w    = (const float*)d_in[3];
  const float* att_b    = (const float*)d_in[4];
  const float* ex_w     = (const float*)d_in[5];
  const float* ex_b     = (const float*)d_in[6];
  const float* qkv_w    = (const float*)d_in[7];
  const float* qkv_b    = (const float*)d_in[8];
  const float* proj_w   = (const float*)d_in[9];
  const float* proj_b   = (const float*)d_in[10];

  char* ws = (char*)d_ws;                      /* all offsets 256B aligned */
  __bf16* xb   = (__bf16*)(ws + 0);            /* 12544*768 bf16 = 19267584 B */
  __bf16* cbb  = (__bf16*)(ws + 19267584);     /* 8192*768 bf16 = 12582912 B */
  float*  cn   = (float*) (ws + 31850496);     /* 8192 f32 */
  float*  vq   = (float*) (ws + 31883264);     /* 12544*768 f32 = 38535168 B */
  float*  mmr  = (float*) (ws + 70418432);     /* 64*196 f32 */
  float*  vd0  = (float*) (ws + 70468608);     /* 64*8*768 f32 */
  float*  qkvb = (float*) (ws + 72041472);     /* 512*2304 f32 */
  float*  ao   = (float*) (ws + 76760064);     /* 512*768 f32 */

  float* outF = (float*)d_out;
  int*   encO = (int*)(outF + (size_t)B_ * R_ * C_);
  float* rmO  = outF + (size_t)B_ * R_ * C_ + (size_t)B_ * L_;

  int nX = B_ * L_ * C_;
  k_cvt_x      <<<(nX + 255) / 256, 256, 0, stream>>>(in_feas, xb, nX);
  k_prep_cb    <<<K_,             256, 0, stream>>>(codebook, cbb, cn);
  k_vq_argmin  <<<B_ * L_ / 32,   256, 0, stream>>>(xb, cbb, cn, codebook, vq, encO);
  k_mm_conv    <<<B_ * L_,        256, 0, stream>>>(in_feas, ex_w, ex_b, mmr);
  k_soft_apply <<<B_,             256, 0, stream>>>(mmr, in_feas, vq);
  k_region_mask<<<B_ * R_,        256, 0, stream>>>(vq, att_w, att_b, rmO);
  k_regions    <<<B_ * R_,        256, 0, stream>>>(vq, rmO, vd0);
  k_qkv        <<<BP_ * SEQ,      256, 0, stream>>>(vd0, qkv_w, qkv_b, qkvb);
  k_attn       <<<BP_ * HEADS,    256, 0, stream>>>(qkvb, ao);
  k_proj       <<<BP_ * SEQ,      256, 0, stream>>>(ao, proj_w, proj_b, outF);
}
